// EncoderLayer_59072980189641
// MI455X (gfx1250) — matrix operand order repaired and checked by reading, not compiled, not run on hardware
//
#include <hip/hip_runtime.h>
#include <math.h>

typedef __bf16 bhalf;
typedef __attribute__((ext_vector_type(16))) __bf16 v16bf;
typedef __attribute__((ext_vector_type(8)))  float  v8f;
typedef __attribute__((ext_vector_type(4)))  unsigned v4u;
typedef __attribute__((ext_vector_type(8)))  int   v8i;
typedef __attribute__((ext_vector_type(4)))  int   v4i;

// ---------------------------------------------------------------------------
// Feature probes (device pass only; host pass falls back automatically)
// ---------------------------------------------------------------------------
#if defined(__has_builtin)
#  if __has_builtin(__builtin_amdgcn_global_load_async_to_lds_b128)
#    define USE_ASYNC_LDS 1
#  endif
#  if __has_builtin(__builtin_amdgcn_tensor_load_to_lds)
#    define USE_TDM 1
#  endif
#  if __has_builtin(__builtin_amdgcn_s_wait_asynccnt)
#    define HAVE_WAIT_ASYNC 1
#  endif
#  if __has_builtin(__builtin_amdgcn_s_wait_tensorcnt)
#    define HAVE_WAIT_TENSOR 1
#  endif
#endif
#ifndef USE_ASYNC_LDS
#define USE_ASYNC_LDS 0
#endif
#ifndef USE_TDM
#define USE_TDM 0
#endif

#if USE_ASYNC_LDS
typedef __attribute__((address_space(1))) v4i gv4i;  // global int4
typedef __attribute__((address_space(3))) v4i lv4i;  // LDS int4

__device__ __forceinline__ void wait_asynccnt0() {
#if defined(HAVE_WAIT_ASYNC)
  __builtin_amdgcn_s_wait_asynccnt(0);
#else
  asm volatile("s_wait_asynccnt 0" ::: "memory");
#endif
}
#endif

#if USE_TDM
__device__ __forceinline__ void wait_tensorcnt0() {
#if defined(HAVE_WAIT_TENSOR)
  __builtin_amdgcn_s_wait_tensorcnt((unsigned short)0);
#else
  asm volatile("s_wait_tensorcnt 0" ::: "memory");
#endif
}
#endif

// ---------------------------------------------------------------------------
// WMMA helper: D = A(16x32 bf16) * B(32x16 bf16) + C(16x16 f32)
// ---------------------------------------------------------------------------
__device__ __forceinline__ v8f wmma_bf16(v16bf a, v16bf b, v8f c) {
  return __builtin_amdgcn_wmma_f32_16x16x32_bf16(
      /*neg_a=*/false, a, /*neg_b=*/false, b,
      /*c_mod=*/(short)0, c, /*reuse_a=*/false, /*reuse_b=*/false);
}

// ---------------------------------------------------------------------------
// Fragment loaders (wave32): lane half h holds k = 16*h + s for both A and B.
// ---------------------------------------------------------------------------
__device__ __forceinline__ v16bf load_frag_a_lds(const bhalf* As, int lda, int mbase) {
  union { v16bf v; unsigned u[8]; } f;
  const int lane = threadIdx.x & 31;
  const int kh = (lane >> 4) & 1;
  const unsigned* row = (const unsigned*)(As + (size_t)(mbase + (lane & 15)) * lda);
#pragma unroll
  for (int v = 0; v < 4; ++v) f.u[v] = row[8 * kh + v];
#pragma unroll
  for (int v = 0; v < 4; ++v) f.u[4 + v] = row[8 * kh + 4 + v];
  return f.v;
}

__device__ __forceinline__ v16bf load_frag_b_lds(const bhalf* Bt, int ldb, int nbase) {
  union { v16bf v; unsigned u[8]; } f;
  const int lane = threadIdx.x & 31;
  const int kh = (lane >> 4) & 1;
  const unsigned* row = (const unsigned*)(Bt + (size_t)(nbase + (lane & 15)) * ldb);
#pragma unroll
  for (int v = 0; v < 8; ++v) f.u[v] = row[8 * kh + v];
  return f.v;
}

// B fragment from a k-major LDS tile Wkn[k][n] (as delivered by the TDM).
__device__ __forceinline__ v16bf load_frag_b_lds_kn(const bhalf* Wkn, int ldn, int nbase) {
  union { v16bf v; bhalf h[16]; } f;
  const int lane = threadIdx.x & 31;
  const int n = nbase + (lane & 15);
  const int k0 = ((lane >> 4) & 1) * 16;
#pragma unroll
  for (int e = 0; e < 16; ++e) f.h[e] = Wkn[(size_t)(k0 + e) * ldn + n];
  return f.v;
}

// Direct-from-global A fragment (rows m0.., K window [k0, k0+32)).
__device__ __forceinline__ v16bf load_frag_a_global(const bhalf* A, int lda, int m0, int k0) {
  union { v16bf v; uint4 q[2]; } f;
  const int lane = threadIdx.x & 31;
  const int kh8 = ((lane >> 4) & 1) * 16;
  const bhalf* row = A + (size_t)(m0 + (lane & 15)) * lda + k0;
  f.q[0] = *(const uint4*)(row + kh8);
  f.q[1] = *(const uint4*)(row + 8 + kh8);
  return f.v;
}

// Direct-from-global B fragment from a row-major B^T buffer: BT[n][k].
__device__ __forceinline__ v16bf load_frag_b_rowmajor(const bhalf* BT, int ldb, int n0, int k0) {
  union { v16bf v; uint4 q[2]; } f;
  const int lane = threadIdx.x & 31;
  const bhalf* row = BT + (size_t)(n0 + (lane & 15)) * ldb + k0 + ((lane >> 4) & 1) * 16;
  f.q[0] = *(const uint4*)(row);
  f.q[1] = *(const uint4*)(row + 8);
  return f.v;
}

// ---------------------------------------------------------------------------
// f32 -> bf16 conversion (grid-stride)
// ---------------------------------------------------------------------------
__global__ void __launch_bounds__(256) cvt_f32_bf16_kernel(
    const float* __restrict__ in, bhalf* __restrict__ out, int n) {
  for (int i = blockIdx.x * blockDim.x + threadIdx.x; i < n;
       i += gridDim.x * blockDim.x)
    out[i] = (bhalf)in[i];
}

// ---------------------------------------------------------------------------
// Generic bf16 WMMA GEMM: C[M,N] = A[M,K] @ W[K,N] + bias, optional relu,
// optional per-head-transposed store for V (mode&2): out[(b,h,dk,s)].
// Block: 128 threads (4 waves), tile 64x64, K-step 32.
// A tile: async global->LDS (ASYNCcnt). W tile: TDM descriptor (TENSORcnt).
// ---------------------------------------------------------------------------
__global__ void __launch_bounds__(128) gemm_bf16_kernel(
    const bhalf* __restrict__ A, const bhalf* __restrict__ W,
    const float* __restrict__ bias, bhalf* __restrict__ C,
    int M, int N, int K, int mode) {
  __shared__ alignas(16) bhalf As[64 * 32];
  __shared__ alignas(16) bhalf Bbuf[64 * 32];  // TDM: [32 k][64 n]; else: [64 n][32 k]

  const int n0 = blockIdx.x * 64;
  const int m0 = blockIdx.y * 64;
  const int tid = threadIdx.x;
  const int wave = tid >> 5;
  const int lane = tid & 31;
  const int l15 = lane & 15;
  const int kh = (lane >> 4) & 1;
  const int wm = (wave >> 1) * 32;
  const int wn = (wave & 1) * 32;

  v8f acc[2][2] = {};

  for (int kb = 0; kb < K; kb += 32) {
    // ---- A tile 64x32 -> LDS (row-major) -------------------------------
    {
      const int row = tid >> 1, seg = tid & 1;
      const bhalf* Ap = A + (size_t)(m0 + row) * K + kb + seg * 16;
#if USE_ASYNC_LDS
      gv4i* gsrc = (gv4i*)Ap;
      lv4i* ldst = (lv4i*)&As[row * 32 + seg * 16];
      __builtin_amdgcn_global_load_async_to_lds_b128(gsrc, ldst, 0, 0);
      __builtin_amdgcn_global_load_async_to_lds_b128(gsrc + 1, ldst + 1, 0, 0);
#else
      uint4 a0 = *(const uint4*)Ap;
      uint4 a1 = *(const uint4*)(Ap + 8);
      *(uint4*)&As[row * 32 + seg * 16] = a0;
      *(uint4*)&As[row * 32 + seg * 16 + 8] = a1;
#endif
    }
    // ---- W tile 32x64 -> LDS -------------------------------------------
#if USE_TDM
    if (wave == 0) {
      // 2D TDM descriptor: tile 32(k) x 64(n), bf16, row-major into Bbuf.
      const unsigned long long ga =
          (unsigned long long)(const char*)(W + (size_t)kb * N + n0);
      const unsigned lds_off =
          (unsigned)(unsigned long long)(__attribute__((address_space(3))) char*)Bbuf;
      v4u g0 = {1u,                               // count=1 user descriptor
                lds_off,                          // lds_addr
                (unsigned)(ga & 0xffffffffu),     // global_addr[31:0]
                (unsigned)((ga >> 32) & 0x01ffffffu) | 0x80000000u};  // [56:32]|type=2
      const int w1 = (int)((N & 0xffff) << 16);           // tensor_dim0[15:0]
      const int w2 = (int)((N >> 16) | ((K & 0xffff) << 16));  // dim0 hi | dim1 lo
      const int w3 = (int)((K >> 16) | (64u << 16));      // dim1 hi | tile_dim0=64
      v8i g1 = {(int)0x00010000,  // workgroup_mask=0, data_size=1 (2B)
                w1, w2, w3,
                32,               // tile_dim1=32, tile_dim2=0
                (int)N,           // tensor_dim0_stride[31:0]
                0, 0};
      v4i g2 = {0, 0, 0, 0};
      v4i g3 = {0, 0, 0, 0};
#if __clang_major__ >= 23
      v8i g4 = {0, 0, 0, 0, 0, 0, 0, 0};
      __builtin_amdgcn_tensor_load_to_lds(g0, g1, g2, g3, g4, 0);
#else
      __builtin_amdgcn_tensor_load_to_lds(g0, g1, g2, g3, 0);
#endif
      wait_tensorcnt0();
    }
#else
    {
      const int kk = tid >> 2, seg = tid & 3;
      const bhalf* Wp = W + (size_t)(kb + kk) * N + n0 + seg * 16;
      union { uint4 q[2]; bhalf h[16]; } tb;
      tb.q[0] = *(const uint4*)Wp;
      tb.q[1] = *(const uint4*)(Wp + 8);
#pragma unroll
      for (int j = 0; j < 16; ++j) Bbuf[(seg * 16 + j) * 32 + kk] = tb.h[j];
    }
#endif
#if USE_ASYNC_LDS
    wait_asynccnt0();
#endif
    __syncthreads();

    v16bf a0 = load_frag_a_lds(As, 32, wm);
    v16bf a1 = load_frag_a_lds(As, 32, wm + 16);
#if USE_TDM
    v16bf b0 = load_frag_b_lds_kn(Bbuf, 64, wn);
    v16bf b1 = load_frag_b_lds_kn(Bbuf, 64, wn + 16);
#else
    v16bf b0 = load_frag_b_lds(Bbuf, 32, wn);
    v16bf b1 = load_frag_b_lds(Bbuf, 32, wn + 16);
#endif
    acc[0][0] = wmma_bf16(a0, b0, acc[0][0]);
    acc[0][1] = wmma_bf16(a0, b1, acc[0][1]);
    acc[1][0] = wmma_bf16(a1, b0, acc[1][0]);
    acc[1][1] = wmma_bf16(a1, b1, acc[1][1]);
    __syncthreads();
  }

#pragma unroll
  for (int im = 0; im < 2; ++im)
#pragma unroll
    for (int jn = 0; jn < 2; ++jn)
#pragma unroll
      for (int r = 0; r < 8; ++r) {
        const int gm = m0 + wm + im * 16 + r + 8 * kh;
        const int gn = n0 + wn + jn * 16 + l15;
        float v = acc[im][jn][r] + bias[gn];
        if (mode & 1) v = fmaxf(v, 0.0f);
        if (mode & 2) {  // V: store as [b][h][dk][s] for attention B-operand
          const int s = gm & 4095, bb = gm >> 12;
          const int hh = gn >> 6, dk = gn & 63;
          C[(((size_t)(bb * 8 + hh) * 64 + dk) << 12) + s] = (bhalf)v;
        } else {
          C[(size_t)gm * N + gn] = (bhalf)v;
        }
      }
}

// ---------------------------------------------------------------------------
// Flash attention: one wave per 16 query rows, 32 keys per iteration.
// Q,K as [B*S, 512] bf16 (head slice via +h*64); Vt as [B][H][64][S] bf16.
// ---------------------------------------------------------------------------
__global__ void __launch_bounds__(128) attention_kernel(
    const bhalf* __restrict__ Q, const bhalf* __restrict__ K,
    const bhalf* __restrict__ Vt, bhalf* __restrict__ AO) {
  const int S = 4096, D = 512;
  const int bh = blockIdx.y;
  const int b = bh >> 3, h = bh & 7;
  const int wave = threadIdx.x >> 5;
  const int lane = threadIdx.x & 31;
  const int l15 = lane & 15;
  const int kh = (lane >> 4) & 1;
  const int q0 = blockIdx.x * 64 + wave * 16;

  const bhalf* Qp = Q + (size_t)b * S * D + h * 64;
  const bhalf* Kp = K + (size_t)b * S * D + h * 64;
  const bhalf* Vp = Vt + (size_t)(b * 8 + h) * 64 * S;

  __shared__ alignas(16) bhalf Pbuf[4][16 * 32];

  const v16bf qa0 = load_frag_a_global(Qp, D, q0, 0);
  const v16bf qa1 = load_frag_a_global(Qp, D, q0, 32);

  v8f acc[4] = {};
  float mrow[8], lrow[8];
#pragma unroll
  for (int r = 0; r < 8; ++r) { mrow[r] = -1e30f; lrow[r] = 0.0f; }

  for (int kt = 0; kt < S; kt += 32) {
    if (kt + 32 < S) {  // stream next K/V tiles toward L2/WGP$
      __builtin_prefetch(Kp + (size_t)(kt + 32) * D, 0, 1);
      __builtin_prefetch(Vp + (kt + 32), 0, 1);
    }
    // S = Q K^T over d_k=64 (two K=32 WMMAs per 16-key tile)
    v8f s0 = {}, s1 = {};
    s0 = wmma_bf16(qa0, load_frag_b_rowmajor(Kp, D, kt, 0), s0);
    s0 = wmma_bf16(qa1, load_frag_b_rowmajor(Kp, D, kt, 32), s0);
    s1 = wmma_bf16(qa0, load_frag_b_rowmajor(Kp, D, kt + 16, 0), s1);
    s1 = wmma_bf16(qa1, load_frag_b_rowmajor(Kp, D, kt + 16, 32), s1);

    float p0[8], p1[8], tmax[8], rsum[8], scale[8];
#pragma unroll
    for (int r = 0; r < 8; ++r) {
      s0[r] *= 0.125f;  // 1/sqrt(64)
      s1[r] *= 0.125f;
      tmax[r] = fmaxf(s0[r], s1[r]);
    }
#pragma unroll
    for (int m = 8; m >= 1; m >>= 1)
#pragma unroll
      for (int r = 0; r < 8; ++r)
        tmax[r] = fmaxf(tmax[r], __shfl_xor(tmax[r], m, 32));

#pragma unroll
    for (int r = 0; r < 8; ++r) {
      const float mn = fmaxf(mrow[r], tmax[r]);
      scale[r] = __expf(mrow[r] - mn);
      mrow[r] = mn;
      p0[r] = __expf(s0[r] - mn);
      p1[r] = __expf(s1[r] - mn);
      rsum[r] = p0[r] + p1[r];
    }
#pragma unroll
    for (int m = 8; m >= 1; m >>= 1)
#pragma unroll
      for (int r = 0; r < 8; ++r) rsum[r] += __shfl_xor(rsum[r], m, 32);
#pragma unroll
    for (int r = 0; r < 8; ++r) lrow[r] = lrow[r] * scale[r] + rsum[r];
#pragma unroll
    for (int cb = 0; cb < 4; ++cb)
#pragma unroll
      for (int r = 0; r < 8; ++r) acc[cb][r] *= scale[r];

    // Re-layout P (C-layout f32) -> A-layout bf16 via LDS
#pragma unroll
    for (int r = 0; r < 8; ++r) {
      const int row = r + 8 * kh;
      Pbuf[wave][row * 32 + l15] = (bhalf)p0[r];
      Pbuf[wave][row * 32 + 16 + l15] = (bhalf)p1[r];
    }
    __syncthreads();
    const v16bf pa = load_frag_a_lds(&Pbuf[wave][0], 32, 0);
#pragma unroll
    for (int cb = 0; cb < 4; ++cb)
      acc[cb] = wmma_bf16(pa, load_frag_b_rowmajor(Vp, S, cb * 16, kt), acc[cb]);
    __syncthreads();
  }

  float rinv[8];
#pragma unroll
  for (int r = 0; r < 8; ++r) rinv[r] = 1.0f / lrow[r];
#pragma unroll
  for (int cb = 0; cb < 4; ++cb)
#pragma unroll
    for (int r = 0; r < 8; ++r) {
      const int row = q0 + r + 8 * kh;
      const int col = h * 64 + cb * 16 + l15;
      AO[((size_t)b * S + row) * D + col] = (bhalf)(acc[cb][r] * rinv[r]);
    }
}

// ---------------------------------------------------------------------------
// Fused residual add + LayerNorm (reference form: (x-mu)/(std+eps)*g+b)
// ---------------------------------------------------------------------------
__device__ __forceinline__ float block_reduce_sum(float v) {
  __shared__ float red[8];
#pragma unroll
  for (int m = 16; m >= 1; m >>= 1) v += __shfl_xor(v, m, 32);
  const int lane = threadIdx.x & 31, wid = threadIdx.x >> 5;
  if (lane == 0) red[wid] = v;
  __syncthreads();
  v = (lane < 8) ? red[lane] : 0.0f;
#pragma unroll
  for (int m = 4; m >= 1; m >>= 1) v += __shfl_xor(v, m, 32);
  v = __shfl(v, 0, 32);
  __syncthreads();
  return v;
}

__global__ void __launch_bounds__(256) add_layernorm_kernel(
    const float* __restrict__ xres, const bhalf* __restrict__ yadd,
    const float* __restrict__ gamma, const float* __restrict__ beta,
    float* __restrict__ outf, bhalf* __restrict__ outb) {
  const size_t base = (size_t)blockIdx.x * 512;
  const int c0 = threadIdx.x, c1 = threadIdx.x + 256;
  const float v0 = xres[base + c0] + (float)yadd[base + c0];
  const float v1 = xres[base + c1] + (float)yadd[base + c1];
  const float mean = block_reduce_sum(v0 + v1) * (1.0f / 512.0f);
  const float d0 = v0 - mean, d1 = v1 - mean;
  const float var = block_reduce_sum(d0 * d0 + d1 * d1) * (1.0f / 512.0f);
  const float rinv = 1.0f / (sqrtf(var) + 1e-5f);
  const float o0 = d0 * rinv * gamma[c0] + beta[c0];
  const float o1 = d1 * rinv * gamma[c1] + beta[c1];
  if (outf) { outf[base + c0] = o0; outf[base + c1] = o1; }
  if (outb) { outb[base + c0] = (bhalf)o0; outb[base + c1] = (bhalf)o1; }
}

// ---------------------------------------------------------------------------
// Host-side orchestration
// ---------------------------------------------------------------------------
extern "C" void kernel_launch(void* const* d_in, const int* in_sizes, int n_in,
                              void* d_out, int out_size, void* d_ws, size_t ws_size,
                              hipStream_t stream) {
  (void)in_sizes; (void)n_in; (void)out_size; (void)ws_size;
  const int S = 4096, D = 512, DFF = 2048, T = 2 * S;

  const float* x  = (const float*)d_in[0];
  const float* Wq = (const float*)d_in[1];  const float* bq = (const float*)d_in[2];
  const float* Wk = (const float*)d_in[3];  const float* bk = (const float*)d_in[4];
  const float* Wv = (const float*)d_in[5];  const float* bv = (const float*)d_in[6];
  const float* Wo = (const float*)d_in[7];  const float* bo = (const float*)d_in[8];
  const float* W1 = (const float*)d_in[9];  const float* b1 = (const float*)d_in[10];
  const float* W2 = (const float*)d_in[11]; const float* b2 = (const float*)d_in[12];
  const float* g1 = (const float*)d_in[13]; const float* be1 = (const float*)d_in[14];
  const float* g2 = (const float*)d_in[15]; const float* be2 = (const float*)d_in[16];
  float* out = (float*)d_out;

  char* base = (char*)d_ws;
  size_t off = 0;
  auto alloc = [&](size_t bytes) -> void* {
    void* p = base + off;
    off += (bytes + 255) & ~(size_t)255;
    return p;
  };

  bhalf* xb  = (bhalf*)alloc((size_t)T * D * 2);
  bhalf* Wqb = (bhalf*)alloc((size_t)D * D * 2);
  bhalf* Wkb = (bhalf*)alloc((size_t)D * D * 2);
  bhalf* Wvb = (bhalf*)alloc((size_t)D * D * 2);
  bhalf* Wob = (bhalf*)alloc((size_t)D * D * 2);
  bhalf* W1b = (bhalf*)alloc((size_t)D * DFF * 2);
  bhalf* W2b = (bhalf*)alloc((size_t)DFF * D * 2);
  bhalf* Qb  = (bhalf*)alloc((size_t)T * D * 2);
  bhalf* Kb  = (bhalf*)alloc((size_t)T * D * 2);
  bhalf* Vt  = (bhalf*)alloc((size_t)T * D * 2);   // [B][H][64][S]
  bhalf* AO  = (bhalf*)alloc((size_t)T * D * 2);
  float* x1f = (float*)alloc((size_t)T * D * 4);
  bhalf* hb  = (bhalf*)alloc((size_t)T * DFF * 2);
  bhalf* proj = Qb;   // reuse: Q dead after attention
  bhalf* x1b  = Kb;   // reuse: K dead after attention
  bhalf* f2   = Vt;   // reuse: Vt dead after FFN input ready

  // f32 -> bf16
  cvt_f32_bf16_kernel<<<1024, 256, 0, stream>>>(x, xb, T * D);
  cvt_f32_bf16_kernel<<<256, 256, 0, stream>>>(Wq, Wqb, D * D);
  cvt_f32_bf16_kernel<<<256, 256, 0, stream>>>(Wk, Wkb, D * D);
  cvt_f32_bf16_kernel<<<256, 256, 0, stream>>>(Wv, Wvb, D * D);
  cvt_f32_bf16_kernel<<<256, 256, 0, stream>>>(Wo, Wob, D * D);
  cvt_f32_bf16_kernel<<<1024, 256, 0, stream>>>(W1, W1b, D * DFF);
  cvt_f32_bf16_kernel<<<1024, 256, 0, stream>>>(W2, W2b, DFF * D);

  // Q/K/V projections
  gemm_bf16_kernel<<<dim3(D / 64, T / 64), 128, 0, stream>>>(xb, Wqb, bq, Qb, T, D, D, 0);
  gemm_bf16_kernel<<<dim3(D / 64, T / 64), 128, 0, stream>>>(xb, Wkb, bk, Kb, T, D, D, 0);
  gemm_bf16_kernel<<<dim3(D / 64, T / 64), 128, 0, stream>>>(xb, Wvb, bv, Vt, T, D, D, 2);

  // Flash attention
  attention_kernel<<<dim3(S / 64, 16), 128, 0, stream>>>(Qb, Kb, Vt, AO);

  // Output projection + LN1
  gemm_bf16_kernel<<<dim3(D / 64, T / 64), 128, 0, stream>>>(AO, Wob, bo, proj, T, D, D, 0);
  add_layernorm_kernel<<<T, 256, 0, stream>>>(x, proj, g1, be1, x1f, x1b);

  // FFN + LN2
  gemm_bf16_kernel<<<dim3(DFF / 64, T / 64), 128, 0, stream>>>(x1b, W1b, b1, hb, T, DFF, D, 1);
  gemm_bf16_kernel<<<dim3(D / 64, T / 64), 128, 0, stream>>>(hb, W2b, b2, f2, T, D, DFF, 0);
  add_layernorm_kernel<<<T, 256, 0, stream>>>(x1f, f2, g2, be2, out, nullptr);
}
